// GptOssExperts_32581621907747
// MI455X (gfx1250) — matrix operand order repaired and checked by reading, not compiled, not run on hardware
//
#include <hip/hip_runtime.h>
#include <hip/hip_bf16.h>

typedef __bf16 bf16;
typedef __attribute__((ext_vector_type(16))) __bf16 v16bf;
typedef __attribute__((ext_vector_type(8)))  __bf16 bf16x8;
typedef __attribute__((ext_vector_type(4)))  __bf16 bf16x4;
typedef __attribute__((ext_vector_type(8)))  float   v8f;
typedef __attribute__((ext_vector_type(4)))  float   f32x4;
typedef __attribute__((ext_vector_type(4)))  int     i32x4;

#define ALPHA_C 1.702f
#define LIMIT_C 7.0f

constexpr int E_    = 8;
constexpr int D_    = 2048;   // hidden dim
constexpr int F_    = 2048;   // ffn dim
constexpr int T_    = 2048;   // tokens (B*S)
constexpr int TWOF_ = 4096;

constexpr int BM  = 128;      // block tile M (tokens)
constexpr int BN1 = 256;      // kernel-1 block tile N (wave tile 64x64)
constexpr int BN2 = 128;      // kernel-2 block tile N (wave tile 64x32)
constexpr int BK  = 32;       // K per LDS stage (one bf16 WMMA k-step)
constexpr int LDSS = 48;      // LDS row stride in bf16 elems (96B, 16B-aligned)

#if __has_builtin(__builtin_amdgcn_global_load_async_to_lds_b128)
#define HAVE_ASYNC_LDS 1
#else
#define HAVE_ASYNC_LDS 0
#endif

#if HAVE_ASYNC_LDS
__device__ inline void async_copy16(const void* g, void* l) {
    __builtin_amdgcn_global_load_async_to_lds_b128(
        (__attribute__((address_space(1))) i32x4*)g,
        (__attribute__((address_space(3))) i32x4*)l, 0, 0);
}
__device__ inline void wait_async0() {
#if __has_builtin(__builtin_amdgcn_s_wait_asynccnt)
    __builtin_amdgcn_s_wait_asynccnt(0);
#else
    asm volatile("s_wait_asynccnt 0x0" ::: "memory");
#endif
}
#endif

__device__ inline v8f zero8() {
    v8f z;
    #pragma unroll
    for (int i = 0; i < 8; ++i) z[i] = 0.0f;
    return z;
}

// A-fragment (16x32 bf16). Lanes 0-15: row m, K {0..7,16..23};
// lanes 16-31: row m, K {8..15,24..31}.  (ISA 7.12.2)
__device__ inline v16bf load_frag_a(const bf16* lds, int row, int lane) {
    const int kb = (lane >> 4) * 8;
    const bf16* p = lds + row * LDSS;
    bf16x8 lo = *(const bf16x8*)(p + kb);
    bf16x8 hi = *(const bf16x8*)(p + kb + 16);
    return __builtin_shufflevector(lo, hi, 0,1,2,3,4,5,6,7,8,9,10,11,12,13,14,15);
}

// B-fragment (32x16 bf16), N-major in LDS. Lanes 0-15: K {0..7,16..23} of col n,
// lanes 16-31: K {8..15,24..31}.
__device__ inline v16bf load_frag_b(const bf16* lds, int col, int lane) {
    const int kb = (lane >> 4) * 8;
    const bf16* p = lds + col * LDSS + kb;
    bf16x8 lo = *(const bf16x8*)(p);
    bf16x8 hi = *(const bf16x8*)(p + 16);
    return __builtin_shufflevector(lo, hi, 0,1,2,3,4,5,6,7,8,9,10,11,12,13,14,15);
}

// ---------------------------------------------------------------------------
// Kernel 0: fp32 -> bf16 convert of hidden_states
// ---------------------------------------------------------------------------
__global__ __launch_bounds__(256) void cvt_x_kernel(const float* __restrict__ src,
                                                    bf16* __restrict__ dst, int n) {
    int i = (blockIdx.x * 256 + threadIdx.x) * 4;
    if (i < n) {
        f32x4 v = *(const f32x4*)(src + i);
        bf16x4 o;
        o[0] = (bf16)v[0]; o[1] = (bf16)v[1]; o[2] = (bf16)v[2]; o[3] = (bf16)v[3];
        *(bf16x4*)(dst + i) = o;
    }
}

// ---------------------------------------------------------------------------
// Kernel 1: gate_up GEMM + bias + clamped GLU -> fused[E][T][F] bf16
// grid: (2F/BN1, T/BM, E), block 256 (8 waves, 2x4), wave tile 64x64
// double-buffered LDS pipeline, async A-tiles
// ---------------------------------------------------------------------------
__global__ __launch_bounds__(256) void gate_up_kernel(const bf16*  __restrict__ xb,
                                                      const float* __restrict__ w,
                                                      const float* __restrict__ bvec,
                                                      bf16*        __restrict__ fused) {
    __shared__ __align__(16) bf16 As[2][BM][LDSS];
    __shared__ __align__(16) bf16 Bs[2][BN1][LDSS];

    const int tid  = threadIdx.x;
    const int lane = tid & 31;
    const int wid  = tid >> 5;
    const int wrow = wid >> 2;        // 0..1  (64-row stripe)
    const int wcol = wid & 3;         // 0..3  (64-col stripe)
    const int e  = blockIdx.z;
    const int m0 = blockIdx.y * BM;
    const int n0 = blockIdx.x * BN1;

    const float* wg = w + (size_t)e * TWOF_ * D_;   // [2F][D] row-major (N x K)

    // per-thread streaming pointers (advance by BK every stage)
    int arow[2], acol[2]; const bf16* aptr[2];
    #pragma unroll
    for (int it = 0; it < 2; ++it) {
        int o = tid + it * 256;                 // 512 x 16B chunks (128x32 bf16)
        arow[it] = o >> 2; acol[it] = (o & 3) * 8;
        aptr[it] = xb + (size_t)(m0 + arow[it]) * D_ + acol[it];
    }
    int brow[8], bcol[8]; const float* bptr[8];
    #pragma unroll
    for (int it = 0; it < 8; ++it) {
        int q = tid + it * 256;                 // 2048 x 16B chunks (256x32 f32)
        brow[it] = q >> 3; bcol[it] = (q & 7) * 4;
        bptr[it] = wg + (size_t)(n0 + brow[it]) * D_ + bcol[it];
    }

    f32x4 breg[8];
    auto stage_a = [&](int buf) {
        #pragma unroll
        for (int it = 0; it < 2; ++it) {
#if HAVE_ASYNC_LDS
            async_copy16(aptr[it], &As[buf][arow[it]][acol[it]]);
#else
            *(bf16x8*)(&As[buf][arow[it]][acol[it]]) = *(const bf16x8*)aptr[it];
#endif
            __builtin_prefetch(aptr[it] + BK, 0, 1);
            aptr[it] += BK;
        }
    };
    auto issue_b = [&]() {
        #pragma unroll
        for (int it = 0; it < 8; ++it) {
            breg[it] = *(const f32x4*)bptr[it];
            __builtin_prefetch(bptr[it] + BK, 0, 1);
            bptr[it] += BK;
        }
    };
    auto commit_b = [&](int buf) {
        #pragma unroll
        for (int it = 0; it < 8; ++it) {
            bf16x4 o4;
            o4[0] = (bf16)breg[it][0]; o4[1] = (bf16)breg[it][1];
            o4[2] = (bf16)breg[it][2]; o4[3] = (bf16)breg[it][3];
            *(bf16x4*)(&Bs[buf][brow[it]][bcol[it]]) = o4;
        }
    };

    v8f acc[4][4];
    #pragma unroll
    for (int i = 0; i < 4; ++i)
        #pragma unroll
        for (int j = 0; j < 4; ++j) acc[i][j] = zero8();

    // prologue: stage k0 = 0 into buffer 0
    stage_a(0); issue_b(); commit_b(0);
#if HAVE_ASYNC_LDS
    wait_async0();
#endif
    __syncthreads();

    int p = 0;
    for (int k0 = 0; k0 < D_; k0 += BK) {
        const bool more = (k0 + BK) < D_;
        if (more) { stage_a(p ^ 1); issue_b(); }

        v16bf af[4], bfr[4];
        #pragma unroll
        for (int i = 0; i < 4; ++i)
            af[i] = load_frag_a(&As[p][0][0], wrow * 64 + i * 16 + (lane & 15), lane);
        #pragma unroll
        for (int j = 0; j < 4; ++j)
            bfr[j] = load_frag_b(&Bs[p][0][0], wcol * 64 + j * 16 + (lane & 15), lane);

        #pragma unroll
        for (int i = 0; i < 4; ++i)
            #pragma unroll
            for (int j = 0; j < 4; ++j)
                acc[i][j] = __builtin_amdgcn_wmma_f32_16x16x32_bf16(
                    false, af[i], false, bfr[j], (short)0, acc[i][j], false, false);

        if (more) commit_b(p ^ 1);
#if HAVE_ASYNC_LDS
        wait_async0();
#endif
        __syncthreads();
        p ^= 1;
    }

    // Epilogue: bias + pair (gate,up) across adjacent lanes + activation
    const int lm = lane & 15;
    #pragma unroll
    for (int j = 0; j < 4; ++j) {
        const int col  = n0 + wcol * 64 + j * 16 + lm;   // column in [0, 2F)
        const float bias = bvec[(size_t)e * TWOF_ + col];
        #pragma unroll
        for (int i = 0; i < 4; ++i) {
            const int rbase = m0 + wrow * 64 + i * 16 + ((lane >> 4) << 3);
            v8f c = acc[i][j];
            #pragma unroll
            for (int r = 0; r < 8; ++r) {
                float v = c[r] + bias;
                float other = __shfl_xor(v, 1, 32);
                if ((lane & 1) == 0) {
                    float g = fminf(v, LIMIT_C);
                    float u = fminf(fmaxf(other, -LIMIT_C), LIMIT_C);
                    float glu = g / (1.0f + __expf(-ALPHA_C * g));
                    float fv = (u + 1.0f) * glu;
                    const int row = rbase + r;
                    fused[((size_t)e * T_ + row) * F_ + (col >> 1)] = (bf16)fv;
                }
            }
        }
    }
}

// ---------------------------------------------------------------------------
// Kernel 2: down GEMM + bias, fused with routing-weight mix over all experts
// grid: (D/BN2, T/BM), block 256; persistent fp32 accumulator -> d_out once
// ---------------------------------------------------------------------------
__global__ __launch_bounds__(256) void down_mix_kernel(const bf16*  __restrict__ fused,
                                                       const float* __restrict__ w,
                                                       const float* __restrict__ bvec,
                                                       const float* __restrict__ rweights,
                                                       float*       __restrict__ out) {
    __shared__ __align__(16) bf16 As[2][BM][LDSS];
    __shared__ __align__(16) bf16 Bs[2][BN2][LDSS];
    __shared__ float rw_lds[BM];

    const int tid  = threadIdx.x;
    const int lane = tid & 31;
    const int wid  = tid >> 5;
    const int wrow = wid >> 2;
    const int wcol = wid & 3;
    const int m0 = blockIdx.y * BM;   // token rows
    const int n0 = blockIdx.x * BN2;  // hidden cols h

    v8f outacc[4][2];
    #pragma unroll
    for (int i = 0; i < 4; ++i)
        #pragma unroll
        for (int j = 0; j < 2; ++j) outacc[i][j] = zero8();

    for (int e = 0; e < E_; ++e) {
        __syncthreads();   // protect rw_lds + LDS buffers vs previous expert
        if (tid < BM) rw_lds[tid] = rweights[(size_t)(m0 + tid) * E_ + e];

        const bf16*  fe = fused + (size_t)e * T_ * F_;   // [T][F] bf16
        const float* wd = w + (size_t)e * D_ * F_;       // [D][F] row-major (N x K)

        int arow[2], acol[2]; const bf16* aptr[2];
        #pragma unroll
        for (int it = 0; it < 2; ++it) {
            int o = tid + it * 256;
            arow[it] = o >> 2; acol[it] = (o & 3) * 8;
            aptr[it] = fe + (size_t)(m0 + arow[it]) * F_ + acol[it];
        }
        int brow[4], bcol[4]; const float* bptr[4];
        #pragma unroll
        for (int it = 0; it < 4; ++it) {
            int q = tid + it * 256;
            brow[it] = q >> 3; bcol[it] = (q & 7) * 4;
            bptr[it] = wd + (size_t)(n0 + brow[it]) * F_ + bcol[it];
        }

        f32x4 breg[4];
        auto stage_a = [&](int buf) {
            #pragma unroll
            for (int it = 0; it < 2; ++it) {
#if HAVE_ASYNC_LDS
                async_copy16(aptr[it], &As[buf][arow[it]][acol[it]]);
#else
                *(bf16x8*)(&As[buf][arow[it]][acol[it]]) = *(const bf16x8*)aptr[it];
#endif
                __builtin_prefetch(aptr[it] + BK, 0, 1);
                aptr[it] += BK;
            }
        };
        auto issue_b = [&]() {
            #pragma unroll
            for (int it = 0; it < 4; ++it) {
                breg[it] = *(const f32x4*)bptr[it];
                __builtin_prefetch(bptr[it] + BK, 0, 1);
                bptr[it] += BK;
            }
        };
        auto commit_b = [&](int buf) {
            #pragma unroll
            for (int it = 0; it < 4; ++it) {
                bf16x4 o4;
                o4[0] = (bf16)breg[it][0]; o4[1] = (bf16)breg[it][1];
                o4[2] = (bf16)breg[it][2]; o4[3] = (bf16)breg[it][3];
                *(bf16x4*)(&Bs[buf][brow[it]][bcol[it]]) = o4;
            }
        };

        v8f acc[4][2];
        #pragma unroll
        for (int i = 0; i < 4; ++i)
            #pragma unroll
            for (int j = 0; j < 2; ++j) acc[i][j] = zero8();

        stage_a(0); issue_b(); commit_b(0);
#if HAVE_ASYNC_LDS
        wait_async0();
#endif
        __syncthreads();

        int p = 0;
        for (int k0 = 0; k0 < F_; k0 += BK) {
            const bool more = (k0 + BK) < F_;
            if (more) { stage_a(p ^ 1); issue_b(); }

            v16bf af[4], bfr[2];
            #pragma unroll
            for (int i = 0; i < 4; ++i)
                af[i] = load_frag_a(&As[p][0][0], wrow * 64 + i * 16 + (lane & 15), lane);
            #pragma unroll
            for (int j = 0; j < 2; ++j)
                bfr[j] = load_frag_b(&Bs[p][0][0], wcol * 32 + j * 16 + (lane & 15), lane);

            #pragma unroll
            for (int i = 0; i < 4; ++i)
                #pragma unroll
                for (int j = 0; j < 2; ++j)
                    acc[i][j] = __builtin_amdgcn_wmma_f32_16x16x32_bf16(
                        false, af[i], false, bfr[j], (short)0, acc[i][j], false, false);

            if (more) commit_b(p ^ 1);
#if HAVE_ASYNC_LDS
            wait_async0();
#endif
            __syncthreads();
            p ^= 1;
        }

        // per-expert epilogue: + bias, scale by routing weight, accumulate
        const int lm = lane & 15;
        #pragma unroll
        for (int j = 0; j < 2; ++j) {
            const int col = n0 + wcol * 32 + j * 16 + lm;
            const float db = bvec[(size_t)e * D_ + col];
            #pragma unroll
            for (int i = 0; i < 4; ++i) {
                const int rl = wrow * 64 + i * 16 + ((lane >> 4) << 3);
                #pragma unroll
                for (int r = 0; r < 8; ++r) {
                    float wv = rw_lds[rl + r];
                    outacc[i][j][r] += wv * (acc[i][j][r] + db);
                }
            }
        }
    }

    // final store (each output element written exactly once)
    const int lm = lane & 15;
    #pragma unroll
    for (int j = 0; j < 2; ++j) {
        const int col = n0 + wcol * 32 + j * 16 + lm;
        #pragma unroll
        for (int i = 0; i < 4; ++i) {
            const int rbase = m0 + wrow * 64 + i * 16 + ((lane >> 4) << 3);
            #pragma unroll
            for (int r = 0; r < 8; ++r)
                out[(size_t)(rbase + r) * D_ + col] = outacc[i][j][r];
        }
    }
}

extern "C" void kernel_launch(void* const* d_in, const int* in_sizes, int n_in,
                              void* d_out, int out_size, void* d_ws, size_t ws_size,
                              hipStream_t stream) {
    const float* hs    = (const float*)d_in[0];   // [B,S,D] fp32
    const float* rw    = (const float*)d_in[1];   // [T,E]   fp32
    const float* gup_w = (const float*)d_in[2];   // [E,2F,D] fp32
    const float* gup_b = (const float*)d_in[3];   // [E,2F]  fp32
    const float* dn_w  = (const float*)d_in[4];   // [E,D,F] fp32
    const float* dn_b  = (const float*)d_in[5];   // [E,D]   fp32
    float* out = (float*)d_out;

    bf16* xb    = (bf16*)d_ws;                                           // 8.4 MB
    bf16* fused = (bf16*)((char*)d_ws + (size_t)T_ * D_ * sizeof(bf16)); // 67 MB

    const int n = T_ * D_;
    cvt_x_kernel<<<n / (256 * 4), 256, 0, stream>>>(hs, xb, n);

    dim3 gb(TWOF_ / BN1, T_ / BM, E_);
    gate_up_kernel<<<gb, 256, 0, stream>>>(xb, gup_w, gup_b, fused);

    dim3 gc(D_ / BN2, T_ / BM);
    down_mix_kernel<<<gc, 256, 0, stream>>>(fused, dn_w, dn_b, rw, out);
}
